// HierarchicalSlots_39015482917300
// MI455X (gfx1250) — matrix operand order repaired and checked by reading, not compiled, not run on hardware
//
#include <hip/hip_runtime.h>
#include <hip/hip_bf16.h>

typedef __attribute__((ext_vector_type(16))) _Float16 v16h;
typedef __attribute__((ext_vector_type(8)))  float    v8f;

#define B_     8
#define N_     4096
#define DF_    768
#define K1_    8
#define K2_    2
#define DS_    128
#define KVQ_   128
#define BK_    64          // B*K1
#define SCALE_ 0.08838834764831845f  // 128^-0.5

// ---------------------------------------------------------------------------
// K0: per-(b,k1) mean of parent_masks over N
// ---------------------------------------------------------------------------
__global__ void hs_denom(const float* __restrict__ masks, float* __restrict__ wdenom) {
    int bk = blockIdx.x, t = threadIdx.x;
    __shared__ float red[256];
    float s = 0.f;
    for (int n = t; n < N_; n += 256) s += masks[bk * N_ + n];
    red[t] = s; __syncthreads();
    for (int o = 128; o > 0; o >>= 1) { if (t < o) red[t] += red[t + o]; __syncthreads(); }
    if (t == 0) wdenom[bk] = red[0] / (float)N_;
}

// ---------------------------------------------------------------------------
// K1: convert Wk, Wv (f32 [128,768]) -> f16 workspace copies
// ---------------------------------------------------------------------------
__global__ void hs_wconv(const float* __restrict__ Wk, const float* __restrict__ Wv,
                         _Float16* __restrict__ wk16, _Float16* __restrict__ wv16) {
    int i = blockIdx.x * blockDim.x + threadIdx.x;   // 0 .. 2*98304-1
    const int M = KVQ_ * DF_;
    if (i < M)            wk16[i]     = (_Float16)Wk[i];
    else if (i < 2 * M)   wv16[i - M] = (_Float16)Wv[i - M];
}

// ---------------------------------------------------------------------------
// K2: init slots = parent_slots + 0.01*eps  (flat [BK,K2,DS] == eps layout)
// ---------------------------------------------------------------------------
__global__ void hs_init_slots(const float* __restrict__ parent_slots,
                              const float* __restrict__ eps,
                              float* __restrict__ slots) {
    int i = blockIdx.x * blockDim.x + threadIdx.x;   // 0..16383
    if (i < BK_ * K2_ * DS_) {
        int d  = i & 127;
        int bk = i >> 8;                             // i / (K2*DS)
        slots[i] = parent_slots[bk * DS_ + d] + 0.01f * eps[i];
    }
}

// ---------------------------------------------------------------------------
// K3: fused gate + LayerNorm + K/V projection via WMMA f32_16x16x32_f16
// grid: (N/32, BK), block: 512 (16 waves). Each block: 32 token rows.
// Waves 0..7 : k-cols [16w,16w+16), both row-halves (2 WMMAs / B-frag)
// Waves 8..15: v-cols [16(w-8),...), both row-halves
// ---------------------------------------------------------------------------
__global__ __launch_bounds__(512)
void hs_projkv(const float* __restrict__ tokens,
               const float* __restrict__ masks,
               const float* __restrict__ wdenom,
               const float* __restrict__ lng,
               const float* __restrict__ lnb,
               const _Float16* __restrict__ wk16,
               const _Float16* __restrict__ wv16,
               _Float16* __restrict__ kbuf,
               _Float16* __restrict__ vbuf) {
    __shared__ _Float16 lds_a[32][DF_ + 8];
    __shared__ float ps[32][16], pq[32][16];
    __shared__ float rowm[32], rowr[32];

    const int t  = threadIdx.x;     // 0..511
    const int bk = blockIdx.y;
    const int n0 = blockIdx.x * 32;
    const int r  = t >> 4;          // row 0..31
    const int g  = t & 15;          // column group
    const int b  = bk >> 3;
    const int n  = n0 + r;

    // gating weight for this row
    const float dn   = wdenom[bk];
    const float mk   = masks[bk * N_ + n];
    const float wrow = (dn > 1e-6f) ? (mk / (dn + 1e-6f)) : 1.0f;

    const float* trow = tokens + ((size_t)b * N_ + n) * DF_;

    // pass 1: row sums of gated values
    float s = 0.f, sq = 0.f;
    #pragma unroll 4
    for (int j = 0; j < DF_ / 16; ++j) {
        float x = trow[g + 16 * j] * wrow;
        s += x; sq += x * x;
    }
    ps[r][g] = s; pq[r][g] = sq;
    __syncthreads();
    if (t < 32) {
        float ss = 0.f, qq = 0.f;
        #pragma unroll
        for (int j = 0; j < 16; ++j) { ss += ps[t][j]; qq += pq[t][j]; }
        float mean = ss / (float)DF_;
        float var  = qq / (float)DF_ - mean * mean;
        rowm[t] = mean;
        rowr[t] = rsqrtf(var + 1e-5f);
    }
    __syncthreads();

    // pass 2: normalize, scale, store f16 A-tile to LDS
    {
        const float mean = rowm[r], rstd = rowr[r];
        #pragma unroll 4
        for (int j = 0; j < DF_ / 16; ++j) {
            int c = g + 16 * j;
            float x = trow[c] * wrow;
            float y = (x - mean) * rstd * lng[c] + lnb[c];
            lds_a[r][c] = (_Float16)y;
        }
    }
    __syncthreads();

    // WMMA phase
    const int wv_  = t >> 5;              // 0..15
    const int lane = t & 31;
    const int lrow = lane & 15;
    const int sel  = lane >> 4;
    const int ct   = wv_ & 7;             // col-tile within the matrix
    const bool isv = (wv_ >= 8);
    const _Float16* __restrict__ W16  = isv ? wv16 : wk16;
    _Float16* __restrict__       obuf = isv ? vbuf : kbuf;
    const int colq = ct * 16 + lrow;      // B-matrix column (output feature)

    v8f c0 = {}; v8f c1 = {};
    #pragma unroll 1
    for (int kk = 0; kk < DF_ / 32; ++kk) {
        const int d0 = kk * 32;
        // A fragments: lane holds row lrow (+16); K set {kb..kb+15}, kb = 16*sel
        v16h a0, a1;
        const _Float16* ar0 = &lds_a[lrow][d0 + sel * 16];
        const _Float16* ar1 = &lds_a[lrow + 16][d0 + sel * 16];
        #pragma unroll
        for (int p = 0; p < 4; ++p) {
            a0[2*p]   = ar0[2*p];        a0[2*p+1]   = ar0[2*p+1];
            a1[2*p]   = ar1[2*p];        a1[2*p+1]   = ar1[2*p+1];
        }
        #pragma unroll
        for (int p = 0; p < 4; ++p) {
            a0[8+2*p] = ar0[8 + 2*p];   a0[8+2*p+1] = ar0[8 + 2*p + 1];
            a1[8+2*p] = ar1[8 + 2*p];   a1[8+2*p+1] = ar1[8 + 2*p + 1];
        }
        // B fragment: lane holds col colq, 16 consecutive K (= d) values; loaded once
        const int dbase = d0 + sel * 16;
        v16h bf = *(const v16h*)(W16 + (size_t)colq * DF_ + dbase);
        c0 = __builtin_amdgcn_wmma_f32_16x16x32_f16(false, a0, false, bf, (short)0, c0, false, false);
        c1 = __builtin_amdgcn_wmma_f32_16x16x32_f16(false, a1, false, bf, (short)0, c1, false, false);
    }

    // D layout: VGPR e -> M = 8*sel + e, N = lane&15
    const int colN  = ct * 16 + (lane & 15);
    const int mbase = sel * 8;
    #pragma unroll
    for (int e = 0; e < 8; ++e) {
        obuf[((size_t)bk * N_ + n0 +      mbase + e) * KVQ_ + colN] = (_Float16)c0[e];
        obuf[((size_t)bk * N_ + n0 + 16 + mbase + e) * KVQ_ + colN] = (_Float16)c1[e];
    }
}

// ---------------------------------------------------------------------------
// K4: slot LN + Q projection; also snapshot slots_prev. grid 64, block 128.
// ---------------------------------------------------------------------------
__global__ void hs_slot_prep(const float* __restrict__ slots,
                             float* __restrict__ slots_prev,
                             const float* __restrict__ lnsg,
                             const float* __restrict__ lnsb,
                             const float* __restrict__ Wq,
                             float* __restrict__ qbuf) {
    __shared__ float sl[2][128], sn[2][128], red[128];
    __shared__ float mv[2];
    const int bk = blockIdx.x, t = threadIdx.x;

    sl[0][t] = slots[bk * 256 + t];
    sl[1][t] = slots[bk * 256 + 128 + t];
    slots_prev[bk * 256 + t]       = sl[0][t];
    slots_prev[bk * 256 + 128 + t] = sl[1][t];
    __syncthreads();

    for (int s = 0; s < 2; ++s) {
        float x = sl[s][t];
        red[t] = x; __syncthreads();
        for (int o = 64; o > 0; o >>= 1) { if (t < o) red[t] += red[t + o]; __syncthreads(); }
        if (t == 0) mv[0] = red[0] / 128.f;
        __syncthreads();
        float dx = x - mv[0];
        red[t] = dx * dx; __syncthreads();
        for (int o = 64; o > 0; o >>= 1) { if (t < o) red[t] += red[t + o]; __syncthreads(); }
        if (t == 0) mv[1] = rsqrtf(red[0] / 128.f + 1e-5f);
        __syncthreads();
        sn[s][t] = dx * mv[1] * lnsg[t] + lnsb[t];
        __syncthreads();
    }

    #pragma unroll
    for (int s = 0; s < 2; ++s) {
        float acc = 0.f;
        const float* wq = Wq + (size_t)t * DS_;
        for (int j = 0; j < DS_; ++j) acc += sn[s][j] * wq[j];
        qbuf[bk * 256 + s * 128 + t] = acc;
    }
}

// ---------------------------------------------------------------------------
// K5: attention: dots, softmax over K2, +eps, N-normalize, weighted V sum.
// grid 64 (bk), block 256. attn (pre-normalization) kept in ws for output.
// ---------------------------------------------------------------------------
__global__ void hs_attention(const _Float16* __restrict__ kbuf,
                             const _Float16* __restrict__ vbuf,
                             const float* __restrict__ qbuf,
                             float* __restrict__ attn,
                             float* __restrict__ updates) {
    __shared__ float qs[2][128];
    __shared__ float attnL[2][N_];
    __shared__ float red[256];
    __shared__ float Sinv[2];
    const int bk = blockIdx.x, t = threadIdx.x;

    qs[t >> 7][t & 127] = qbuf[bk * 256 + t];
    __syncthreads();

    float s0 = 0.f, s1 = 0.f;
    for (int n = t; n < N_; n += 256) {
        const v16h* kr16 = (const v16h*)(kbuf + ((size_t)bk * N_ + n) * KVQ_);
        float d0 = 0.f, d1 = 0.f;
        #pragma unroll
        for (int c = 0; c < KVQ_ / 16; ++c) {
            v16h kv = kr16[c];
            #pragma unroll
            for (int e = 0; e < 16; ++e) {
                float f = (float)kv[e];
                d0 += qs[0][c * 16 + e] * f;
                d1 += qs[1][c * 16 + e] * f;
            }
        }
        d0 *= SCALE_; d1 *= SCALE_;
        float mx = fmaxf(d0, d1);
        float e0 = expf(d0 - mx), e1 = expf(d1 - mx);
        float inv = 1.f / (e0 + e1);
        float a0 = e0 * inv + 1e-8f;
        float a1 = e1 * inv + 1e-8f;
        attnL[0][n] = a0; attnL[1][n] = a1;
        attn[(size_t)bk * 2 * N_ + n]      = a0;
        attn[(size_t)bk * 2 * N_ + N_ + n] = a1;
        s0 += a0; s1 += a1;
    }
    red[t] = s0; __syncthreads();
    for (int o = 128; o > 0; o >>= 1) { if (t < o) red[t] += red[t + o]; __syncthreads(); }
    if (t == 0) Sinv[0] = 1.f / red[0];
    __syncthreads();
    red[t] = s1; __syncthreads();
    for (int o = 128; o > 0; o >>= 1) { if (t < o) red[t] += red[t + o]; __syncthreads(); }
    if (t == 0) Sinv[1] = 1.f / red[0];
    __syncthreads();

    // updates[s][d] = sum_n attn_w[s][n] * v[n][d]
    const int s = t >> 7, d = t & 127;
    const float inv = Sinv[s];
    const _Float16* vb = vbuf + (size_t)bk * N_ * KVQ_ + d;
    float acc = 0.f;
    for (int n = 0; n < N_; ++n)
        acc += attnL[s][n] * (float)vb[(size_t)n * KVQ_];
    updates[(bk * 2 + s) * KVQ_ + d] = acc * inv;
}

// ---------------------------------------------------------------------------
// K6: GRU cell + LN + MLP (gelu). grid 128 (slot row), block 128.
// ---------------------------------------------------------------------------
__global__ void hs_gru_mlp(const float* __restrict__ updates,
                           const float* __restrict__ slots_prev,
                           float* __restrict__ slots,
                           const float* __restrict__ Wih, const float* __restrict__ Whh,
                           const float* __restrict__ bih, const float* __restrict__ bhh,
                           const float* __restrict__ mlng, const float* __restrict__ mlnb,
                           const float* __restrict__ W1, const float* __restrict__ b1,
                           const float* __restrict__ W2, const float* __restrict__ b2) {
    __shared__ float x[128], h[128], h2[128], y1[512], red[128];
    __shared__ float mv[2];
    const int r = blockIdx.x, t = threadIdx.x;

    x[t] = updates[r * 128 + t];
    h[t] = slots_prev[r * 128 + t];
    __syncthreads();

    float ir = bih[t], iz = bih[128 + t], in_ = bih[256 + t];
    float hr = bhh[t], hz = bhh[128 + t], hn  = bhh[256 + t];
    const float* wi0 = Wih + (size_t)t * 128;
    const float* wi1 = Wih + (size_t)(128 + t) * 128;
    const float* wi2 = Wih + (size_t)(256 + t) * 128;
    const float* wh0 = Whh + (size_t)t * 128;
    const float* wh1 = Whh + (size_t)(128 + t) * 128;
    const float* wh2 = Whh + (size_t)(256 + t) * 128;
    for (int d = 0; d < 128; ++d) {
        float xd = x[d], hd = h[d];
        ir += xd * wi0[d]; iz += xd * wi1[d]; in_ += xd * wi2[d];
        hr += hd * wh0[d]; hz += hd * wh1[d]; hn  += hd * wh2[d];
    }
    float rg = 1.f / (1.f + expf(-(ir + hr)));
    float z  = 1.f / (1.f + expf(-(iz + hz)));
    float nn = tanhf(in_ + rg * hn);
    float s1 = (1.f - z) * nn + z * h[t];

    // LayerNorm
    red[t] = s1; __syncthreads();
    for (int o = 64; o > 0; o >>= 1) { if (t < o) red[t] += red[t + o]; __syncthreads(); }
    if (t == 0) mv[0] = red[0] / 128.f;
    __syncthreads();
    float dx = s1 - mv[0];
    red[t] = dx * dx; __syncthreads();
    for (int o = 64; o > 0; o >>= 1) { if (t < o) red[t] += red[t + o]; __syncthreads(); }
    if (t == 0) mv[1] = rsqrtf(red[0] / 128.f + 1e-5f);
    __syncthreads();
    h2[t] = dx * mv[1] * mlng[t] + mlnb[t];
    __syncthreads();

    // MLP up + exact gelu
    #pragma unroll
    for (int jj = 0; jj < 4; ++jj) {
        int j = t + 128 * jj;
        float acc = b1[j];
        const float* w = W1 + (size_t)j * 128;
        for (int d = 0; d < 128; ++d) acc += h2[d] * w[d];
        y1[j] = 0.5f * acc * (1.f + erff(acc * 0.70710678118654752f));
    }
    __syncthreads();

    // MLP down
    float acc = b2[t];
    const float* w = W2 + (size_t)t * 512;
    for (int j = 0; j < 512; ++j) acc += y1[j] * w[j];
    slots[r * 128 + t] = acc;
}

// ---------------------------------------------------------------------------
// K7: write outputs: child_objects (16384) then child_gating (524288)
// ---------------------------------------------------------------------------
__global__ void hs_finalize(const float* __restrict__ slots,
                            const float* __restrict__ attn,
                            const float* __restrict__ masks,
                            float* __restrict__ out) {
    int i = blockIdx.x * blockDim.x + threadIdx.x;
    if (i < BK_ * K2_ * DS_) out[i] = slots[i];
    if (i < BK_ * K2_ * N_) {
        int n  = i & (N_ - 1);
        int bk = i >> 13;                 // / (K2*N)
        out[BK_ * K2_ * DS_ + i] = attn[i] * masks[bk * N_ + n];
    }
}

// ---------------------------------------------------------------------------
extern "C" void kernel_launch(void* const* d_in, const int* in_sizes, int n_in,
                              void* d_out, int out_size, void* d_ws, size_t ws_size,
                              hipStream_t stream) {
    (void)in_sizes; (void)n_in; (void)out_size; (void)ws_size;

    const float* tokens       = (const float*)d_in[0];
    const float* parent_slots = (const float*)d_in[1];
    const float* parent_masks = (const float*)d_in[2];
    const float* eps_noise    = (const float*)d_in[3];
    const float* ln_in_g      = (const float*)d_in[4];
    const float* ln_in_b      = (const float*)d_in[5];
    const float* ln_s_g       = (const float*)d_in[6];
    const float* ln_s_b       = (const float*)d_in[7];
    const float* Wq           = (const float*)d_in[8];
    const float* Wk           = (const float*)d_in[9];
    const float* Wv           = (const float*)d_in[10];
    const float* gru_Wih      = (const float*)d_in[11];
    const float* gru_Whh      = (const float*)d_in[12];
    const float* gru_bih      = (const float*)d_in[13];
    const float* gru_bhh      = (const float*)d_in[14];
    const float* mlp_ln_g     = (const float*)d_in[15];
    const float* mlp_ln_b     = (const float*)d_in[16];
    const float* mlp_W1       = (const float*)d_in[17];
    const float* mlp_b1       = (const float*)d_in[18];
    const float* mlp_W2       = (const float*)d_in[19];
    const float* mlp_b2       = (const float*)d_in[20];
    float* out = (float*)d_out;

    // workspace carve-up (256B aligned)
    char* base = (char*)d_ws;
    size_t off = 0;
    auto take = [&](size_t bytes) -> void* {
        void* p = base + off;
        off = (off + bytes + 255) & ~(size_t)255;
        return p;
    };
    _Float16* wk16    = (_Float16*)take((size_t)KVQ_ * DF_ * sizeof(_Float16));
    _Float16* wv16    = (_Float16*)take((size_t)KVQ_ * DF_ * sizeof(_Float16));
    float* wdenom     = (float*)take(BK_ * sizeof(float));
    float* slots      = (float*)take((size_t)BK_ * K2_ * DS_ * sizeof(float));
    float* slots_prev = (float*)take((size_t)BK_ * K2_ * DS_ * sizeof(float));
    float* qbuf       = (float*)take((size_t)BK_ * K2_ * KVQ_ * sizeof(float));
    float* updates    = (float*)take((size_t)BK_ * K2_ * KVQ_ * sizeof(float));
    float* attn       = (float*)take((size_t)BK_ * K2_ * N_ * sizeof(float));
    _Float16* kbuf    = (_Float16*)take((size_t)BK_ * N_ * KVQ_ * sizeof(_Float16));
    _Float16* vbuf    = (_Float16*)take((size_t)BK_ * N_ * KVQ_ * sizeof(_Float16));

    // prologue
    hs_denom<<<BK_, 256, 0, stream>>>(parent_masks, wdenom);
    hs_wconv<<<(2 * KVQ_ * DF_ + 255) / 256, 256, 0, stream>>>(Wk, Wv, wk16, wv16);
    hs_init_slots<<<(BK_ * K2_ * DS_ + 255) / 256, 256, 0, stream>>>(parent_slots, eps_noise, slots);

    // fused gate+LN+K/V projection (WMMA)
    hs_projkv<<<dim3(N_ / 32, BK_), 512, 0, stream>>>(tokens, parent_masks, wdenom,
                                                      ln_in_g, ln_in_b, wk16, wv16, kbuf, vbuf);

    // 3 slot-attention iterations
    for (int it = 0; it < 3; ++it) {
        hs_slot_prep<<<BK_, 128, 0, stream>>>(slots, slots_prev, ln_s_g, ln_s_b, Wq, qbuf);
        hs_attention<<<BK_, 256, 0, stream>>>(kbuf, vbuf, qbuf, attn, updates);
        hs_gru_mlp<<<BK_ * K2_, 128, 0, stream>>>(updates, slots_prev, slots,
                                                  gru_Wih, gru_Whh, gru_bih, gru_bhh,
                                                  mlp_ln_g, mlp_ln_b, mlp_W1, mlp_b1, mlp_W2, mlp_b2);
    }

    // outputs
    hs_finalize<<<(BK_ * K2_ * N_ + 255) / 256, 256, 0, stream>>>(slots, attn, parent_masks, out);
}
